// Group_Attention_67138928771093
// MI455X (gfx1250) — matrix operand order repaired and checked by reading, not compiled, not run on hardware
//
#include <hip/hip_runtime.h>
#include <hip/hip_bf16.h>

// ---------------------------------------------------------------------------
// Grouped attention (B=8, N=4096, C=512, H=8, Dh=64, groups=8 -> L=512) on
// MI455X/gfx1250. All matmul stages on v_wmma_f32_16x16x32_bf16, softmax and
// accumulation fp32. ~103 GFLOP, ~0.45 GB HBM traffic -> ~20us floor @ 23.3
// TB/s; bf16 WMMA keeps compute in the same order.
//
// Round-2 changes vs round-1 (driven by disasm):
//  * qkv GEMM splits Q/K (normal layout) from V (stored transposed V^T[d][key],
//    key-contiguous). C-layout accumulators make the transposed store a single
//    b128 per 8 values; attention V B-frags become 2x global_load_b128 instead
//    of 16 strided u16 loads + 16 v_mov_b16 packs.
//  * Row sums via an extra WMMA against an all-ones B tile (exact, rescalable)
//    instead of 32 ds_bpermute per key-step.
//  * Running softmax max is shared per 16-lane half (covers that half's 8
//    rows): 4 ds_bpermute per key-step instead of 64 total.
// Workspace (~170 MB): x_bf16 | WqkvT | WprojT | qk_bf16 | vT_bf16 | attn_out.
// ---------------------------------------------------------------------------

typedef __bf16 bf16_t;
typedef __attribute__((ext_vector_type(16))) __bf16    v16bf;
typedef __attribute__((ext_vector_type(8)))  float     v8f;
typedef __attribute__((ext_vector_type(4)))  unsigned  u32x4;
typedef __attribute__((ext_vector_type(2)))  unsigned  u32x2;
typedef __attribute__((ext_vector_type(4)))  float     f32x4;

union FragU { u32x4 u[2]; v16bf v; };

__device__ __forceinline__ v8f zero_v8f() {
    v8f z;
#pragma unroll
    for (int i = 0; i < 8; ++i) z[i] = 0.0f;
    return z;
}

__device__ __forceinline__ v8f wmma_bf16(v16bf a, v16bf b, v8f c) {
    // D = A(16x32 bf16) * B(32x16 bf16) + C(16x16 f32)
    return __builtin_amdgcn_wmma_f32_16x16x32_bf16(
        /*neg_a=*/false, a, /*neg_b=*/false, b,
        /*c_mod=*/(short)0, c, /*reuse_a=*/false, /*reuse_b=*/false);
}

// A-fragment (16x32 bf16): lane row m = lane&15; lanes<16 hold K={0..7,16..23},
// lanes>=16 hold K={8..15,24..31}.  Two 16B loads.
__device__ __forceinline__ v16bf load_frag_A(const bf16_t* A, int lda,
                                             int m0, int k0, int lane) {
    int m  = m0 + (lane & 15);
    int kb = ((lane >> 4) & 1) * 8;
    const bf16_t* p = A + (size_t)m * lda + k0 + kb;
    FragU f;
    f.u[0] = *reinterpret_cast<const u32x4*>(p);        // K = kb .. kb+7
    f.u[1] = *reinterpret_cast<const u32x4*>(p + 16);   // K = 16+kb .. 16+kb+7
    return f.v;
}

// B-fragment (32x16 bf16) from an n-major, k-contiguous matrix: lane col
// n = lane&15; lanes<16 hold K={0..7,16..23}, lanes>=16 hold K={8..15,24..31}. Two 16B loads.
__device__ __forceinline__ v16bf load_frag_B(const bf16_t* Bt, int ldb,
                                             int n0, int k0, int lane) {
    int n  = n0 + (lane & 15);
    int kk = k0 + ((lane >> 4) & 1) * 8;
    const bf16_t* p = Bt + (size_t)n * ldb + kk;
    FragU f;
    f.u[0] = *reinterpret_cast<const u32x4*>(p);        // K = kk .. kk+7
    f.u[1] = *reinterpret_cast<const u32x4*>(p + 16);   // K = kk+16 .. kk+23
    return f.v;
}

__device__ __forceinline__ u32x4 pack8_bf16(v8f a) {
    union { bf16_t h[8]; u32x4 u; } p;
#pragma unroll
    for (int i = 0; i < 8; ++i) p.h[i] = (bf16_t)a[i];
    return p.u;
}

// ---------------------------------------------------------------------------
// Conversion kernels
// ---------------------------------------------------------------------------
__global__ __launch_bounds__(256) void cvt_bf16_vec4(const float* __restrict__ in,
                                                     bf16_t* __restrict__ out, int n) {
    int i = (blockIdx.x * blockDim.x + threadIdx.x) * 4;
    if (i >= n) return;
    f32x4 f = *reinterpret_cast<const f32x4*>(in + i);
    union { bf16_t h[4]; u32x2 u; } p;
#pragma unroll
    for (int j = 0; j < 4; ++j) p.h[j] = (bf16_t)f[j];
    *reinterpret_cast<u32x2*>(out + i) = p.u;
}

// W[K][N] (fp32, row-major) -> Wt[N][K] (bf16) so B-frags are contiguous.
__global__ __launch_bounds__(256) void cvt_transpose_bf16(const float* __restrict__ W,
                                                          bf16_t* __restrict__ Wt,
                                                          int K, int N) {
    int idx = blockIdx.x * blockDim.x + threadIdx.x;
    if (idx >= K * N) return;
    int k = idx / N;
    int n = idx - k * N;
    Wt[(size_t)n * K + k] = (bf16_t)W[idx];
}

// ---------------------------------------------------------------------------
// GEMM core: block = 256 threads = 8 waves tiled 4(M) x 2(N); wave tile 32x32;
// block tile 128x64; K stepped by 32.  A[M][K] bf16, Bt[N][K] bf16.
// ---------------------------------------------------------------------------
#define GEMM_CORE(A, Bt, K)                                                   \
    int lane = threadIdx.x & 31;                                              \
    int wave = threadIdx.x >> 5;                                              \
    int wm = wave & 3;                                                        \
    int wn = wave >> 2;                                                       \
    int m0 = blockIdx.x * 128 + wm * 32;                                      \
    int n0 = blockIdx.y * 64  + wn * 32;                                      \
    v8f acc[2][2];                                                            \
    acc[0][0] = zero_v8f(); acc[0][1] = zero_v8f();                           \
    acc[1][0] = zero_v8f(); acc[1][1] = zero_v8f();                           \
    for (int k0 = 0; k0 < (K); k0 += 32) {                                    \
        v16bf a0 = load_frag_A((A), (K), m0,      k0, lane);                  \
        v16bf a1 = load_frag_A((A), (K), m0 + 16, k0, lane);                  \
        v16bf b0 = load_frag_B((Bt), (K), n0,      k0, lane);                 \
        v16bf b1 = load_frag_B((Bt), (K), n0 + 16, k0, lane);                 \
        acc[0][0] = wmma_bf16(a0, b0, acc[0][0]);                             \
        acc[0][1] = wmma_bf16(a0, b1, acc[0][1]);                             \
        acc[1][0] = wmma_bf16(a1, b0, acc[1][0]);                             \
        acc[1][1] = wmma_bf16(a1, b1, acc[1][1]);                             \
    }                                                                         \
    int nl = lane & 15;                                                       \
    int mh = (lane >> 4) * 8;

// Q/K part of qkv (+ generic GEMM): normal [m][n] output, bf16 or f32+bias.
template <bool OUT_BF16>
__global__ __launch_bounds__(256) void gemm_bf16(const bf16_t* __restrict__ A,
                                                 const bf16_t* __restrict__ Bt,
                                                 bf16_t* __restrict__ Dbf,
                                                 float*  __restrict__ Df,
                                                 const float* __restrict__ bias,
                                                 int K, int ldd) {
    GEMM_CORE(A, Bt, K)
#pragma unroll
    for (int ti = 0; ti < 2; ++ti)
#pragma unroll
        for (int tj = 0; tj < 2; ++tj)
#pragma unroll
            for (int i = 0; i < 8; ++i) {
                int mm = m0 + ti * 16 + mh + i;
                int nn = n0 + tj * 16 + nl;
                float v = acc[ti][tj][i];
                if (OUT_BF16)
                    Dbf[(size_t)mm * ldd + nn] = (bf16_t)v;
                else
                    Df[(size_t)mm * ldd + nn] = v + bias[nn];
            }
}

// V part of qkv: output column nv = h*64+d, row m = grp*512+key; store
// transposed as vt[grp*512 + nv][key] (V^T, key-contiguous) -- one b128 per
// 8 accumulator values since i indexes consecutive m (= keys).
__global__ __launch_bounds__(256) void gemm_qkv_v(const bf16_t* __restrict__ A,
                                                  const bf16_t* __restrict__ Bt,
                                                  bf16_t* __restrict__ vt,
                                                  int K) {
    GEMM_CORE(A, Bt, K)
#pragma unroll
    for (int ti = 0; ti < 2; ++ti)
#pragma unroll
        for (int tj = 0; tj < 2; ++tj) {
            int nv  = n0 + tj * 16 + nl;          // h*64 + d
            int mb  = m0 + ti * 16 + mh;          // grp*512 + key0 (8 keys)
            int grp = mb >> 9;
            int key = mb & 511;
            size_t row = (size_t)grp * 512 + nv;
            *reinterpret_cast<u32x4*>(vt + row * 512 + key) =
                pack8_bf16(acc[ti][tj]);
        }
}

// ---------------------------------------------------------------------------
// Flash attention per (b,g,h): Q,K from qk[32768][1024] (offsets h*64 and
// 512+h*64), V from vt (V^T[64][512] per head, key-contiguous). One block per
// head (512 blocks), 8 waves; each wave owns 4 query tiles of 16 rows and
// streams keys in tiles of 32 with online softmax.
// Row sums accumulate through WMMA against an all-ones B tile; the running
// max is shared per 16-lane half (covers that half's 8 rows).
// ---------------------------------------------------------------------------
__global__ __launch_bounds__(256) void attn_fwd(const bf16_t* __restrict__ qk,
                                                const bf16_t* __restrict__ vt,
                                                bf16_t* __restrict__ aout) {
    const int ROW = 1024;
    int bid = blockIdx.x;                 // b*64 + g*8 + h
    int h   = bid & 7;
    int grp = bid >> 3;                   // b*8 + g
    const bf16_t* base = qk + (size_t)grp * 512 * ROW;
    const bf16_t* Q  = base + h * 64;
    const bf16_t* Kp = base + 512 + h * 64;
    const bf16_t* Vb = vt + (size_t)(grp * 512 + h * 64) * 512;  // V^T, ld=512

    int lane = threadIdx.x & 31;
    int wave = threadIdx.x >> 5;
    __shared__ alignas(16) bf16_t plds[8][16][32];   // per-wave P repack tile

    const float scale = 0.125f;           // Dh^-0.5, Dh=64

    v16bf ones;
#pragma unroll
    for (int j = 0; j < 16; ++j) ones[j] = (bf16_t)1.0f;

    for (int t = 0; t < 4; ++t) {
        int m0 = (wave * 4 + t) * 16;
        v16bf qa0 = load_frag_A(Q, ROW, m0, 0,  lane);
        v16bf qa1 = load_frag_A(Q, ROW, m0, 32, lane);

        v8f o[4];
#pragma unroll
        for (int nt = 0; nt < 4; ++nt) o[nt] = zero_v8f();
        v8f lacc = zero_v8f();            // per-row sums (C layout, exact)
        float mcur = -1e30f;              // running max, shared per lane-half

        for (int kb = 0; kb < 512; kb += 32) {
            // S = Q * K^T for 32 keys: two 16x16 score tiles, K-dim = Dh = 64.
            v8f s0 = zero_v8f(), s1 = zero_v8f();
            s0 = wmma_bf16(qa0, load_frag_B(Kp, ROW, kb,      0,  lane), s0);
            s0 = wmma_bf16(qa1, load_frag_B(Kp, ROW, kb,      32, lane), s0);
            s1 = wmma_bf16(qa0, load_frag_B(Kp, ROW, kb + 16, 0,  lane), s1);
            s1 = wmma_bf16(qa1, load_frag_B(Kp, ROW, kb + 16, 32, lane), s1);

            float v0[8], v1[8];
            float tmax = -1e30f;
#pragma unroll
            for (int i = 0; i < 8; ++i) {
                v0[i] = s0[i] * scale;
                v1[i] = s1[i] * scale;
                tmax = fmaxf(tmax, fmaxf(v0[i], v1[i]));
            }
            // half-wide max: masks 1..8 stay within each 16-lane half, which
            // owns rows {0..7} or {8..15} at all 16 key columns.
#pragma unroll
            for (int msk = 1; msk <= 8; msk <<= 1)
                tmax = fmaxf(tmax, __shfl_xor(tmax, msk, 32));

            float mn = fmaxf(mcur, tmax);
            float sc = __expf(mcur - mn);
            mcur = mn;
#pragma unroll
            for (int i = 0; i < 8; ++i) {
                lacc[i] *= sc;
                o[0][i] *= sc; o[1][i] *= sc; o[2][i] *= sc; o[3][i] *= sc;
            }

            // exp probs -> per-wave LDS (C layout -> A layout repack).
            {
                int c  = lane & 15;
                int r0 = (lane >> 4) * 8;
#pragma unroll
                for (int i = 0; i < 8; ++i) {
                    plds[wave][r0 + i][c]      = (bf16_t)__expf(v0[i] - mn);
                    plds[wave][r0 + i][16 + c] = (bf16_t)__expf(v1[i] - mn);
                }
            }
            FragU pf;
            {
                int mr = lane & 15;
                int cc = (lane >> 4) * 8;
                pf.u[0] = *reinterpret_cast<const u32x4*>(&plds[wave][mr][cc]);
                pf.u[1] = *reinterpret_cast<const u32x4*>(&plds[wave][mr][16 + cc]);
            }
            v16bf pa = pf.v;

            // Row sums: lacc += P(16x32) @ ones(32x16)  (every column = rowsum)
            lacc = wmma_bf16(pa, ones, lacc);

            // O += P(16x32) * V(32x64): V^T rows are d, key-contiguous.
#pragma unroll
            for (int nt = 0; nt < 4; ++nt)
                o[nt] = wmma_bf16(pa, load_frag_B(Vb, 512, nt * 16, kb, lane),
                                  o[nt]);
        }

        // Normalize and store bf16 at [grp*512 + m][h*64 + d], row stride 512.
        int nl = lane & 15;
        int mh = (lane >> 4) * 8;
        size_t outbase = (size_t)grp * 512 * 512 + h * 64;
#pragma unroll
        for (int nt = 0; nt < 4; ++nt) {
#pragma unroll
            for (int i = 0; i < 8; ++i) {
                int m = m0 + mh + i;
                float v = o[nt][i] / lacc[i];
                aout[outbase + (size_t)m * 512 + nt * 16 + nl] = (bf16_t)v;
            }
        }
    }
}

// ---------------------------------------------------------------------------
// Host launcher
// ---------------------------------------------------------------------------
extern "C" void kernel_launch(void* const* d_in, const int* in_sizes, int n_in,
                              void* d_out, int out_size, void* d_ws, size_t ws_size,
                              hipStream_t stream) {
    (void)in_sizes; (void)n_in; (void)out_size; (void)ws_size;
    const float* x     = (const float*)d_in[0];   // [8,4096,512]
    const float* Wqkv  = (const float*)d_in[1];   // [512,1536]
    const float* Wproj = (const float*)d_in[2];   // [512,512]
    const float* bproj = (const float*)d_in[3];   // [512]
    // d_in[4] = recursive_index, static 0 -> num_groups = 8.

    char* ws = (char*)d_ws;
    size_t off = 0;
    bf16_t* xb     = (bf16_t*)(ws + off); off += (size_t)8 * 4096 * 512  * 2;  // 33.5 MB
    bf16_t* wqkvT  = (bf16_t*)(ws + off); off += (size_t)1536 * 512      * 2;  //  1.6 MB
    bf16_t* wprojT = (bf16_t*)(ws + off); off += (size_t)512 * 512       * 2;  //  0.5 MB
    bf16_t* qkb    = (bf16_t*)(ws + off); off += (size_t)8 * 4096 * 1024 * 2;  // 67.1 MB
    bf16_t* vtb    = (bf16_t*)(ws + off); off += (size_t)8 * 4096 * 512  * 2;  // 33.5 MB
    bf16_t* aoutb  = (bf16_t*)(ws + off); off += (size_t)8 * 4096 * 512  * 2;  // 33.5 MB

    const int M = 8 * 4096;   // 32768 rows

    // 1) fp32 -> bf16 conversions (weights transposed to k-contiguous rows).
    int nX = M * 512;
    cvt_bf16_vec4<<<nX / 4 / 256, 256, 0, stream>>>(x, xb, nX);
    cvt_transpose_bf16<<<(512 * 1536 + 255) / 256, 256, 0, stream>>>(Wqkv, wqkvT, 512, 1536);
    cvt_transpose_bf16<<<(512 * 512 + 255) / 256, 256, 0, stream>>>(Wproj, wprojT, 512, 512);

    // 2a) Q,K = x @ W_qkv[:, :1024] -> bf16 [32768,1024]
    gemm_bf16<true><<<dim3(M / 128, 1024 / 64), 256, 0, stream>>>(
        xb, wqkvT, qkb, nullptr, nullptr, 512, 1024);
    // 2b) V = x @ W_qkv[:, 1024:] stored transposed: vt[grp*512+h*64+d][key]
    gemm_qkv_v<<<dim3(M / 128, 512 / 64), 256, 0, stream>>>(
        xb, wqkvT + (size_t)1024 * 512, vtb, 512);

    // 3) grouped multi-head flash attention: 512 (b,g,h) blocks.
    attn_fwd<<<8 * 8 * 8, 256, 0, stream>>>(qkb, vtb, aoutb);

    // 4) out = attn_out @ W_proj + b_proj : fp32 [32768,512]
    gemm_bf16<false><<<dim3(M / 128, 512 / 64), 256, 0, stream>>>(
        aoutb, wprojT, nullptr, (float*)d_out, bproj, 512, 512);
}
